// ChatGenerator_5866925326360
// MI455X (gfx1250) — compile-verified, matrix operand order checked by reading, not run on hardware
//
#include <hip/hip_runtime.h>
#include <math.h>
#include <stdint.h>

// Model dims (fixed by the reference)
#define BB   4
#define SS   512
#define DDIM 1024
#define HH   16
#define LLAY 6
#define FFD  4096
#define VV   32000
#define DKK  64

typedef __attribute__((ext_vector_type(16))) __bf16 v16bf;
typedef __attribute__((ext_vector_type(2)))  __bf16 bf2;
typedef __attribute__((ext_vector_type(8)))  float  v8f;

// ---------------------------------------------------------------------------
// WMMA helpers: V_WMMA_F32_16X16X32_BF16, D(16x16 f32) = A(16x32) * B(32x16) + C
// A-fragment layout (16-bit, 16x32, wave32), per CDNA5 ISA:
//   lane = l, m = l&15, hi = l>>4
//   element e = 2*v + j  holds K = ((v>>2)<<4) + (hi<<3) + ((v&3)<<1) + j
// B-fragment mirrors A with N (column) in place of M.
// C/D layout: N = lane&15, VGPR r holds M = r + 8*(lane>>4).
// ---------------------------------------------------------------------------
__device__ __forceinline__ v8f wmma_bf16(v16bf a, v16bf b, v8f c) {
  return __builtin_amdgcn_wmma_f32_16x16x32_bf16(false, a, false, b, (short)0, c,
                                                 false, false);
}

__device__ __forceinline__ int frag_kbase(int v, int hi) {
  return ((v >> 2) << 4) | (hi << 3) | ((v & 3) << 1);
}

// Fragment from fp32 rows with contiguous K (global or LDS), cvt -> bf16.
__device__ __forceinline__ v16bf load_frag_f32(const float* base,
                                               size_t row_stride, int kk) {
  int lane = (int)(threadIdx.x & 31);
  int hi = lane >> 4;
  const float* row = base + (size_t)(lane & 15) * row_stride + kk;
  v16bf f;
#pragma unroll
  for (int v = 0; v < 8; ++v) {
    float2 t = *reinterpret_cast<const float2*>(row + frag_kbase(v, hi));
    f[2 * v]     = (__bf16)t.x;
    f[2 * v + 1] = (__bf16)t.y;
  }
  return f;
}

// B fragment from fp32 K x N storage (K-major rows): element (kc, n).
__device__ __forceinline__ v16bf load_frag_kxn_f32(const float* B, int ldb,
                                                   int n0) {
  int lane = (int)(threadIdx.x & 31);
  int hi = lane >> 4;
  int n = n0 + (lane & 15);
  v16bf f;
#pragma unroll
  for (int v = 0; v < 8; ++v) {
    int kb = frag_kbase(v, hi);
    f[2 * v]     = (__bf16)B[(size_t)(kb)     * ldb + n];
    f[2 * v + 1] = (__bf16)B[(size_t)(kb + 1) * ldb + n];
  }
  return f;
}

// ---------------------------------------------------------------------------
// gfx1250 async global->LDS copy (16B per lane), ASYNCcnt-tracked.
// the LDS address is taken from VGPR vdst; generic LDS pointers carry the LDS byte
// offset in their low 32 bits, so truncation gives the vdst value.
// ---------------------------------------------------------------------------
__device__ __forceinline__ void async_copy_b128(void* lds_ptr, const float* g) {
  uint32_t lds_off = (uint32_t)(uintptr_t)lds_ptr;
  uint64_t gaddr   = (uint64_t)(uintptr_t)g;
  asm volatile("global_load_async_to_lds_b128 %0, %1, off"
               :: "v"(lds_off), "v"(gaddr)
               : "memory");
}

__device__ __forceinline__ void wait_async_zero() {
  asm volatile("s_wait_asynccnt 0x0" ::: "memory");
}

// ---------------------------------------------------------------------------
// Embedding gather + positional encoding (pe row indexed by BATCH — reference
// quirk). One float4 per thread.
// ---------------------------------------------------------------------------
__global__ __launch_bounds__(256) void embed_kernel(const int* __restrict__ ids,
                                                    const float* __restrict__ emb,
                                                    const float* __restrict__ pe,
                                                    float* __restrict__ x) {
  int row = blockIdx.x;              // 0 .. B*S-1
  int b = row / SS;
  int id = ids[row];
  int c = (int)threadIdx.x * 4;      // 256 threads * 4 = D
  float4 e = *reinterpret_cast<const float4*>(emb + (size_t)id * DDIM + c);
  float4 p = *reinterpret_cast<const float4*>(pe + (size_t)b * DDIM + c);
  float4 r;
  r.x = e.x + p.x; r.y = e.y + p.y; r.z = e.z + p.z; r.w = e.w + p.w;
  *reinterpret_cast<float4*>(x + (size_t)row * DDIM + c) = r;
}

// ---------------------------------------------------------------------------
// LayerNorm over D=1024, one row per 256-thread block, one float4 per thread.
// ---------------------------------------------------------------------------
__global__ __launch_bounds__(256) void layernorm_kernel(const float* __restrict__ x,
                                                        const float* __restrict__ g,
                                                        const float* __restrict__ bta,
                                                        float* __restrict__ y) {
  __shared__ float s_sum[8];
  __shared__ float s_sq[8];
  int row = blockIdx.x;
  int c = (int)threadIdx.x * 4;
  const float* xr = x + (size_t)row * DDIM;
  float4 t = *reinterpret_cast<const float4*>(xr + c);
  float sum = t.x + t.y + t.z + t.w;
  float sq  = t.x * t.x + t.y * t.y + t.z * t.z + t.w * t.w;
#pragma unroll
  for (int off = 16; off >= 1; off >>= 1) {
    sum += __shfl_xor(sum, off, 32);
    sq  += __shfl_xor(sq,  off, 32);
  }
  int wid = (int)(threadIdx.x >> 5);
  if ((threadIdx.x & 31) == 0) { s_sum[wid] = sum; s_sq[wid] = sq; }
  __syncthreads();
  if (threadIdx.x < 32) {
    sum = (threadIdx.x < 8) ? s_sum[threadIdx.x] : 0.0f;
    sq  = (threadIdx.x < 8) ? s_sq[threadIdx.x]  : 0.0f;
#pragma unroll
    for (int off = 4; off >= 1; off >>= 1) {
      sum += __shfl_xor(sum, off, 32);
      sq  += __shfl_xor(sq,  off, 32);
    }
    if (threadIdx.x == 0) { s_sum[0] = sum; s_sq[0] = sq; }
  }
  __syncthreads();
  float mean = s_sum[0] * (1.0f / DDIM);
  float var  = s_sq[0] * (1.0f / DDIM) - mean * mean;
  float inv  = rsqrtf(var + 1e-5f);
  float4 gg = *reinterpret_cast<const float4*>(g + c);
  float4 bb = *reinterpret_cast<const float4*>(bta + c);
  float4 r;
  r.x = (t.x - mean) * inv * gg.x + bb.x;
  r.y = (t.y - mean) * inv * gg.y + bb.y;
  r.z = (t.z - mean) * inv * gg.z + bb.z;
  r.w = (t.w - mean) * inv * gg.w + bb.w;
  *reinterpret_cast<float4*>(y + (size_t)row * DDIM + c) = r;
}

// ---------------------------------------------------------------------------
// bf16-WMMA GEMM with async-LDS staging:  C = f(A @ B + bias) (+ residual)
//   A: M x K row-major fp32
//   B: TRANSB ? N x K row-major : K x N row-major, fp32
//   ACT: 0 = identity, 1 = exact GELU
// Block tile 128(M) x 128(N), K-step 32, 8 waves; each wave owns a 32x64
// register tile = 2x4 grid of 16x16 WMMA accumulators (8 WMMAs / K-step).
// Raw fp32 K-slices are staged into LDS with GLOBAL_LOAD_ASYNC_TO_LDS_B128
// (ASYNCcnt + s_wait_asynccnt), and converted fp32->bf16 during fragment
// loads (v_cvt_pk_bf16_f32 co-executes with WMMA).
// Requires M%128==0, N%128==0, K%32==0 (true for every call in this model).
// ---------------------------------------------------------------------------
#define GBM 128
#define GBN 128
#define GBK 32
#define LDSA 36    // fp32 row stride (32 + 4 pad, keeps 16B-aligned rows)
#define LDSBT 36   // TRANSB: B^T rows [n][k]
#define LDSBN 132  // !TRANSB: B rows  [k][n] (128 + 4 pad)

template <int TRANSB, int ACT>
__global__ __launch_bounds__(256) void gemm_wmma_kernel(const float* __restrict__ A,
                                                        const float* __restrict__ B,
                                                        const float* __restrict__ bias,
                                                        const float* __restrict__ residual,
                                                        float* __restrict__ C,
                                                        int M, int N, int K) {
  __shared__ __attribute__((aligned(16))) float sAf[GBM * LDSA];
  __shared__ __attribute__((aligned(16))) float
      sBf[TRANSB ? (GBN * LDSBT) : (GBK * LDSBN)];

  const int t    = (int)threadIdx.x;   // 0..255
  const int wave = t >> 5;             // 0..7
  const int lane = t & 31;
  const int wm = wave >> 1;            // wave M index (0..3) -> 32 rows
  const int wn = wave & 1;             // wave N index (0..1) -> 64 cols
  const int m0 = blockIdx.y * GBM;
  const int n0 = blockIdx.x * GBN;

  v8f zero = {};
  v8f acc[2][4] = {{zero, zero, zero, zero}, {zero, zero, zero, zero}};

  for (int kk = 0; kk < K; kk += GBK) {
    __syncthreads();   // previous iteration's readers done before overwrite
    // ---- stage A: 128 x 32 fp32 (1024 x 16B chunks / 256 threads) ----
#pragma unroll
    for (int i = 0; i < 4; ++i) {
      int f = t + i * 256;
      int r = f >> 3;
      int c = (f & 7) << 2;
      async_copy_b128(&sAf[r * LDSA + c], A + (size_t)(m0 + r) * K + kk + c);
    }
    // ---- stage B ----
    if (TRANSB) {
      // B is N x K: rows contiguous along K -> sBf[n][k]
#pragma unroll
      for (int i = 0; i < 4; ++i) {
        int f = t + i * 256;
        int n = f >> 3;
        int c = (f & 7) << 2;
        async_copy_b128(&sBf[n * LDSBT + c], B + (size_t)(n0 + n) * K + kk + c);
      }
    } else {
      // B is K x N: rows contiguous along N -> sBf[k][n]
#pragma unroll
      for (int i = 0; i < 4; ++i) {
        int f = t + i * 256;
        int kc = f >> 5;
        int n  = (f & 31) << 2;
        async_copy_b128(&sBf[kc * LDSBN + n], B + (size_t)(kk + kc) * N + n0 + n);
      }
    }
    wait_async_zero();   // this wave's LDS writes visible
    __syncthreads();     // all waves' writes visible

    // ---- compute: 2 A frags x 4 B frags -> 8 WMMAs per wave ----
    v16bf afr[2];
#pragma unroll
    for (int i = 0; i < 2; ++i)
      afr[i] = load_frag_f32(&sAf[(wm * 32 + i * 16) * LDSA], LDSA, 0);
    v16bf bfr[4];
#pragma unroll
    for (int j = 0; j < 4; ++j) {
      if (TRANSB)
        bfr[j] = load_frag_f32(&sBf[(wn * 64 + j * 16) * LDSBT], LDSBT, 0);
      else
        bfr[j] = load_frag_kxn_f32(&sBf[0], LDSBN, wn * 64 + j * 16);
    }
#pragma unroll
    for (int i = 0; i < 2; ++i)
#pragma unroll
      for (int j = 0; j < 4; ++j)
        acc[i][j] = wmma_bf16(afr[i], bfr[j], acc[i][j]);
  }

  // ---- epilogue ----
  const int hi = lane >> 4;
  const int ln = lane & 15;
#pragma unroll
  for (int i = 0; i < 2; ++i) {
#pragma unroll
    for (int j = 0; j < 4; ++j) {
      int n = n0 + wn * 64 + j * 16 + ln;
      float bv = bias ? bias[n] : 0.0f;
#pragma unroll
      for (int r = 0; r < 8; ++r) {
        int m = m0 + wm * 32 + i * 16 + r + hi * 8;
        float xv = acc[i][j][r] + bv;
        if (ACT == 1) xv = 0.5f * xv * (1.0f + erff(xv * 0.70710678118f));
        if (residual) xv += residual[(size_t)m * N + n];
        C[(size_t)m * N + n] = xv;
      }
    }
  }
}

// ---------------------------------------------------------------------------
// Causal attention, one (batch, head, 16-query tile) per wave.
// q,k,v,o are [B, S, H, DK] fp32 (seq row stride = D).
// Phase 1: 16x512 masked/scaled QK^T score strip via WMMA -> LDS.
// Phase 2: row softmax in LDS (wave shuffles).
// Phase 3: O(16x64) = P @ V via WMMA, P fragments read from LDS.
// ---------------------------------------------------------------------------
__global__ __launch_bounds__(32) void attn_kernel(const float* __restrict__ q,
                                                  const float* __restrict__ k,
                                                  const float* __restrict__ v,
                                                  float* __restrict__ o) {
  __shared__ float sc[16][SS];     // 32 KB of the 320 KB WGP LDS
  int qt = blockIdx.x;
  int bh = blockIdx.y;
  int b = bh / HH;
  int h = bh % HH;
  int s0 = qt * 16;
  int lane = (int)threadIdx.x;
  int hi = lane >> 4;
  int ln = lane & 15;
  const float scale = 0.125f;      // 1/sqrt(DK=64)

  const float* qbase = q + ((size_t)(b * SS + s0) * HH + h) * DKK;
  v16bf aq0 = load_frag_f32(qbase, (size_t)DDIM, 0);
  v16bf aq1 = load_frag_f32(qbase, (size_t)DDIM, 32);

  // Phase 1: scores
  for (int n0 = 0; n0 < SS; n0 += 16) {
    const float* kb = k + ((size_t)(b * SS + n0) * HH + h) * DKK;
    v16bf bk0 = load_frag_f32(kb, (size_t)DDIM, 0);   // K^T rows contiguous
    v16bf bk1 = load_frag_f32(kb, (size_t)DDIM, 32);
    v8f c = {};
    c = wmma_bf16(aq0, bk0, c);
    c = wmma_bf16(aq1, bk1, c);
#pragma unroll
    for (int r = 0; r < 8; ++r) {
      int m = r + hi * 8;
      int qs = s0 + m;
      int ks = n0 + ln;
      sc[m][n0 + ln] = (ks <= qs) ? c[r] * scale : -INFINITY;
    }
  }
  __syncthreads();

  // Phase 2: row softmax
  for (int r = 0; r < 16; ++r) {
    float mx = -INFINITY;
    for (int ci = lane; ci < SS; ci += 32) mx = fmaxf(mx, sc[r][ci]);
#pragma unroll
    for (int off = 16; off >= 1; off >>= 1) mx = fmaxf(mx, __shfl_xor(mx, off, 32));
    float sum = 0.0f;
    for (int ci = lane; ci < SS; ci += 32) {
      float e = __expf(sc[r][ci] - mx);
      sc[r][ci] = e;
      sum += e;
    }
#pragma unroll
    for (int off = 16; off >= 1; off >>= 1) sum += __shfl_xor(sum, off, 32);
    float inv = 1.0f / sum;
    for (int ci = lane; ci < SS; ci += 32) sc[r][ci] *= inv;
  }
  __syncthreads();

  // Phase 3: O = P @ V
  v8f zero = {};
  v8f oa[4] = {zero, zero, zero, zero};
  for (int n0 = 0; n0 < SS; n0 += 32) {
    v16bf ap = load_frag_f32(&sc[0][0] + n0, (size_t)SS, 0);
#pragma unroll
    for (int c4 = 0; c4 < 4; ++c4) {
      // V fragment: element (kc, n) at v[b, n0+kc, h, c4*16+n]; per-kc stride D
      const float* vb = v + ((size_t)(b * SS + n0) * HH + h) * DKK + c4 * 16;
      v16bf bv = load_frag_kxn_f32(vb, DDIM, 0);
      oa[c4] = wmma_bf16(ap, bv, oa[c4]);
    }
  }
#pragma unroll
  for (int c4 = 0; c4 < 4; ++c4) {
#pragma unroll
    for (int r = 0; r < 8; ++r) {
      int m = r + hi * 8;
      o[((size_t)(b * SS + s0 + m) * HH + h) * DKK + c4 * 16 + ln] = oa[c4][r];
    }
  }
}

// ---------------------------------------------------------------------------
// Host orchestration
// ---------------------------------------------------------------------------
extern "C" void kernel_launch(void* const* d_in, const int* in_sizes, int n_in,
                              void* d_out, int out_size, void* d_ws, size_t ws_size,
                              hipStream_t stream) {
  (void)in_sizes; (void)n_in; (void)out_size; (void)ws_size;

  const int*   ids   = (const int*)d_in[0];
  const float* emb   = (const float*)d_in[1];
  const float* pe    = (const float*)d_in[2];
  const float* Wq    = (const float*)d_in[3];
  const float* Wk    = (const float*)d_in[4];
  const float* Wv    = (const float*)d_in[5];
  const float* Wo    = (const float*)d_in[6];
  const float* bo    = (const float*)d_in[7];
  const float* ln1_s = (const float*)d_in[8];
  const float* ln1_b = (const float*)d_in[9];
  const float* ln2_s = (const float*)d_in[10];
  const float* ln2_b = (const float*)d_in[11];
  const float* W1    = (const float*)d_in[12];
  const float* b1    = (const float*)d_in[13];
  const float* W2    = (const float*)d_in[14];
  const float* b2    = (const float*)d_in[15];
  const float* lnf_s = (const float*)d_in[16];
  const float* lnf_b = (const float*)d_in[17];
  float* out = (float*)d_out;

  const size_t ND = (size_t)BB * SS * DDIM;   // 2048 * 1024
  float* ws = (float*)d_ws;
  float* x  = ws;            // [B*S, D] residual stream
  float* hn = x  + ND;       // [B*S, D] LN output
  float* qb = hn + ND;       // [B*S, D] == [B,S,H,DK]
  float* kb = qb + ND;
  float* vb = kb + ND;
  float* ob = vb + ND;
  float* ff = ob + ND;       // [B*S, FF]

  const int rows = BB * SS;  // 2048

  embed_kernel<<<rows, 256, 0, stream>>>(ids, emb, pe, x);

  dim3 blk(256);
  dim3 gD (DDIM / GBN, rows / GBM);   // 8   x 16
  dim3 gFF(FFD  / GBN, rows / GBM);   // 32  x 16
  dim3 gV (VV   / GBN, rows / GBM);   // 250 x 16
  dim3 gAtt(SS / 16, BB * HH);

  for (int i = 0; i < LLAY; ++i) {
    const float* wq = Wq + (size_t)i * DDIM * DDIM;
    const float* wk = Wk + (size_t)i * DDIM * DDIM;
    const float* wv = Wv + (size_t)i * DDIM * DDIM;
    const float* wo = Wo + (size_t)i * DDIM * DDIM;
    const float* w1 = W1 + (size_t)i * DDIM * FFD;
    const float* w2 = W2 + (size_t)i * FFD * DDIM;

    layernorm_kernel<<<rows, 256, 0, stream>>>(x, ln1_s + i * DDIM, ln1_b + i * DDIM, hn);
    gemm_wmma_kernel<0, 0><<<gD, blk, 0, stream>>>(hn, wq, nullptr, nullptr, qb, rows, DDIM, DDIM);
    gemm_wmma_kernel<0, 0><<<gD, blk, 0, stream>>>(hn, wk, nullptr, nullptr, kb, rows, DDIM, DDIM);
    gemm_wmma_kernel<0, 0><<<gD, blk, 0, stream>>>(hn, wv, nullptr, nullptr, vb, rows, DDIM, DDIM);
    attn_kernel<<<gAtt, 32, 0, stream>>>(qb, kb, vb, ob);
    // x = x + o @ Wo + bo
    gemm_wmma_kernel<0, 0><<<gD, blk, 0, stream>>>(ob, wo, bo + i * DDIM, x, x, rows, DDIM, DDIM);
    layernorm_kernel<<<rows, 256, 0, stream>>>(x, ln2_s + i * DDIM, ln2_b + i * DDIM, hn);
    // ff = gelu(hn @ W1 + b1)
    gemm_wmma_kernel<0, 1><<<gFF, blk, 0, stream>>>(hn, w1, b1 + i * FFD, nullptr, ff, rows, FFD, DDIM);
    // x = x + ff @ W2 + b2
    gemm_wmma_kernel<0, 0><<<gD, blk, 0, stream>>>(ff, w2, b2 + i * DDIM, x, x, rows, DDIM, FFD);
  }

  layernorm_kernel<<<rows, 256, 0, stream>>>(x, lnf_s, lnf_b, hn);
  // logits = hn @ emb.T  (emb is V x D row-major -> TRANSB)
  gemm_wmma_kernel<1, 0><<<gV, blk, 0, stream>>>(hn, emb, nullptr, nullptr, out, rows, VV, DDIM);
}
